// Attention_67388036874796
// MI455X (gfx1250) — matrix operand order repaired and checked by reading, not compiled, not run on hardware
//
#include <hip/hip_runtime.h>
#include <hip/hip_bf16.h>

// CDNA5 (gfx1250) flash-attention forward, causal, fp32 I/O, bf16 WMMA compute.
// Wave32; v_wmma_f32_16x16x32_bf16 for QK^T (computed transposed) and PV.
// Software-pipelined: tile t+1 global loads overlap tile t WMMA/softmax.

typedef __attribute__((ext_vector_type(16))) __bf16 v16bf;
typedef __attribute__((ext_vector_type(8)))  float  v8f;

#define B_SZ   2
#define H_SZ   16
#define S_LEN  2048
#define D_DIM  128
#define QTILE  128               // q rows per block (8 waves x 16)
#define KVTILE 32                // kv rows per LDS tile
#define KS_STRIDE (D_DIM + 8)    // 136 bf16 -> conflict-free row reads
#define VT_STRIDE (KVTILE + 8)   // 40 bf16  -> conflict-free col reads

union BF16Vec {
    v16bf  v;
    uint4  u[2];
    __bf16 e[16];
};
union BF16x8  { __bf16 e[8]; uint4 u; };
union BF16x2  { __bf16 h[2]; unsigned int u; };

__global__ __launch_bounds__(256)
void fa_fwd_cdna5(const float* __restrict__ qg, const float* __restrict__ kg,
                  const float* __restrict__ vg, float* __restrict__ og)
{
    __shared__ __bf16 Ks[KVTILE * KS_STRIDE];  // [kv][d], row-major bf16
    __shared__ __bf16 Vt[D_DIM * VT_STRIDE];   // [d][kv], transposed bf16

    const int tid  = threadIdx.x;
    const int wave = tid >> 5;
    const int lane = tid & 31;
    const int ln16 = lane & 15;
    const int half = lane >> 4;

    const int    q0   = blockIdx.x * QTILE;
    const size_t base = (size_t)blockIdx.y * S_LEN * D_DIM;  // (b,h) offset
    const int    q0w  = q0 + wave * 16;                      // wave's first q row
    const int    qrow = q0w + ln16;                          // lane's q (S^T layout)

    // ---- Q rows -> bf16 B-layout regs, pre-scaled by 1/sqrt(D) ----
    // B operand (32x16): lane n holds column n%16; this code's k order: half 0 -> K 0..15, half 1 -> K 16..31.
    const float rscale = 0.08838834764831845f;  // 1/sqrt(128)
    BF16Vec qb[4];
    {
        const float* qp = qg + base + (size_t)qrow * D_DIM;
        #pragma unroll
        for (int c = 0; c < 4; ++c) {
            const float4* p4 = reinterpret_cast<const float4*>(qp + 32 * c + 16 * half);
            #pragma unroll
            for (int i = 0; i < 4; ++i) {
                float4 f = p4[i];
                qb[c].e[4*i+0] = (__bf16)(f.x * rscale);
                qb[c].e[4*i+1] = (__bf16)(f.y * rscale);
                qb[c].e[4*i+2] = (__bf16)(f.z * rscale);
                qb[c].e[4*i+3] = (__bf16)(f.w * rscale);
            }
        }
    }

    v8f acc[8];
    #pragma unroll
    for (int j = 0; j < 8; ++j) acc[j] = (v8f)(0.0f);
    float mrun = -INFINITY;
    float lrun = 0.0f;

    const int ntiles = q0 / KVTILE + QTILE / KVTILE;  // up to block diagonal

    // Cooperative loader: each thread owns 2 kv rows x 8 d elements.
    const int kp2 = (tid >> 4) * 2;      // kv row pair: {kp2, kp2+1}
    const int d0  = (tid & 15) * 8;      // d offset (8 floats)

    float4 kf[4], vf[4];                 // reg double-buffer (next tile)
    auto load_regs = [&](int t) {
        const size_t roff = base + (size_t)(t * KVTILE + kp2) * D_DIM + d0;
        const float4* kp4 = reinterpret_cast<const float4*>(kg + roff);
        const float4* vp4 = reinterpret_cast<const float4*>(vg + roff);
        kf[0] = kp4[0]; kf[1] = kp4[1];
        kf[2] = kp4[D_DIM / 4]; kf[3] = kp4[D_DIM / 4 + 1];   // row kp2+1
        vf[0] = vp4[0]; vf[1] = vp4[1];
        vf[2] = vp4[D_DIM / 4]; vf[3] = vp4[D_DIM / 4 + 1];
    };

    load_regs(0);

    for (int t = 0; t < ntiles; ++t) {
        const int kv0 = t * KVTILE;

        // ---- convert reg buffer -> bf16 LDS tiles ----
        {
            float ka[16], va[16];        // [0..7]=row kp2, [8..15]=row kp2+1
            #pragma unroll
            for (int i = 0; i < 4; ++i) {
                ka[4*i+0] = kf[i].x; ka[4*i+1] = kf[i].y;
                ka[4*i+2] = kf[i].z; ka[4*i+3] = kf[i].w;
                va[4*i+0] = vf[i].x; va[4*i+1] = vf[i].y;
                va[4*i+2] = vf[i].z; va[4*i+3] = vf[i].w;
            }
            BF16x8 r0, r1;
            #pragma unroll
            for (int i = 0; i < 8; ++i) { r0.e[i] = (__bf16)ka[i]; r1.e[i] = (__bf16)ka[8+i]; }
            *reinterpret_cast<uint4*>(&Ks[(kp2    ) * KS_STRIDE + d0]) = r0.u;
            *reinterpret_cast<uint4*>(&Ks[(kp2 + 1) * KS_STRIDE + d0]) = r1.u;
            #pragma unroll
            for (int i = 0; i < 8; ++i) {   // V transposed, 2 k per b32 store
                BF16x2 pu;
                pu.h[0] = (__bf16)va[i];
                pu.h[1] = (__bf16)va[8+i];
                *reinterpret_cast<unsigned int*>(&Vt[(d0 + i) * VT_STRIDE + kp2]) = pu.u;
            }
        }
        __syncthreads();

        // ---- issue next tile's global loads; they overlap the compute below ----
        if (t + 1 < ntiles) load_regs(t + 1);

        // ---- S^T = K_tile * Q^T : two 16(kv)x16(q) tiles, 4 chained K=32 WMMAs each ----
        v8f st0 = (v8f)(0.0f), st1 = (v8f)(0.0f);
        #pragma unroll
        for (int c = 0; c < 4; ++c) {
            BF16Vec a;
            const int ro = ln16 * KS_STRIDE + 32 * c + 16 * half;
            a.u[0] = *reinterpret_cast<const uint4*>(&Ks[ro]);
            a.u[1] = *reinterpret_cast<const uint4*>(&Ks[ro + 8]);
            st0 = __builtin_amdgcn_wmma_f32_16x16x32_bf16(false, a.v, false, qb[c].v,
                                                          (short)0, st0, false, false);
        }
        #pragma unroll
        for (int c = 0; c < 4; ++c) {
            BF16Vec a;
            const int ro = (16 + ln16) * KS_STRIDE + 32 * c + 16 * half;
            a.u[0] = *reinterpret_cast<const uint4*>(&Ks[ro]);
            a.u[1] = *reinterpret_cast<const uint4*>(&Ks[ro + 8]);
            st1 = __builtin_amdgcn_wmma_f32_16x16x32_bf16(false, a.v, false, qb[c].v,
                                                          (short)0, st1, false, false);
        }

        // ---- online softmax; lane L holds 16 kv scores for q = q0w + L%16 ----
        float p[16];
        #pragma unroll
        for (int r = 0; r < 8; ++r) { p[r] = st0[r]; p[8 + r] = st1[r]; }

        // Causal mask only on diagonal tiles (wave-uniform -> scalar branch)
        if (kv0 + KVTILE - 1 > q0w) {
            #pragma unroll
            for (int r = 0; r < 8; ++r) {
                const int kvA = kv0 + r + 8 * half;   // C-layout: M = r + 8*half
                if (kvA      > qrow) p[r]     = -INFINITY;
                if (kvA + 16 > qrow) p[8 + r] = -INFINITY;
            }
        }

        float tmax = -INFINITY;
        #pragma unroll
        for (int i = 0; i < 16; ++i) tmax = fmaxf(tmax, p[i]);
        tmax = fmaxf(tmax, __shfl_xor(tmax, 16, 32));   // pair lanes share q
        const float mnew  = fmaxf(mrun, tmax);
        const float alpha = __expf(mrun - mnew);
        float rsum = 0.0f;
        #pragma unroll
        for (int i = 0; i < 16; ++i) { p[i] = __expf(p[i] - mnew); rsum += p[i]; }
        rsum += __shfl_xor(rsum, 16, 32);
        lrun = lrun * alpha + rsum;
        mrun = mnew;

        // rescale accumulators: acc row M=r+8*half needs alpha held by lane (r+8*half)
        #pragma unroll
        for (int r = 0; r < 8; ++r) {
            const float ar = __shfl(alpha, r + 8 * half, 32);
            #pragma unroll
            for (int j = 0; j < 8; ++j) acc[j][r] *= ar;
        }

        // ---- P matches bf16 A-layout directly: elems 0..7 = st0 regs, 8..15 = st1 ----
        BF16Vec pa;
        #pragma unroll
        for (int i = 0; i < 16; ++i) pa.e[i] = (__bf16)p[i];

        // ---- out += P(16x32) * V(32x128): 8 column tiles, one WMMA each ----
        #pragma unroll
        for (int j = 0; j < 8; ++j) {
            BF16Vec vb;
            const int ro = (16 * j + ln16) * VT_STRIDE + 8 * half;
            vb.u[0] = *reinterpret_cast<const uint4*>(&Vt[ro]);
            vb.u[1] = *reinterpret_cast<const uint4*>(&Vt[ro + 16]);
            acc[j] = __builtin_amdgcn_wmma_f32_16x16x32_bf16(false, pa.v, false, vb.v,
                                                             (short)0, acc[j], false, false);
        }
        __syncthreads();
    }

    // ---- epilogue: divide by l (per-q, via bpermute) and store fp32 ----
    #pragma unroll
    for (int r = 0; r < 8; ++r) {
        const float lr  = __shfl(lrun, r + 8 * half, 32);
        const float inv = 1.0f / lr;
        const int   qr  = q0w + r + 8 * half;
        float* op = og + base + (size_t)qr * D_DIM + ln16;
        #pragma unroll
        for (int j = 0; j < 8; ++j)
            op[16 * j] = acc[j][r] * inv;   // lanes 0..15 -> 64B coalesced runs
    }
}

extern "C" void kernel_launch(void* const* d_in, const int* in_sizes, int n_in,
                              void* d_out, int out_size, void* d_ws, size_t ws_size,
                              hipStream_t stream) {
    const float* q = (const float*)d_in[0];
    const float* k = (const float*)d_in[1];
    const float* v = (const float*)d_in[2];
    float* o = (float*)d_out;
    dim3 grid(S_LEN / QTILE, B_SZ * H_SZ);   // (16, 32) blocks
    fa_fwd_cdna5<<<grid, 256, 0, stream>>>(q, k, v, o);
}
